// Attention_7687991460097
// MI455X (gfx1250) — matrix operand order repaired and checked by reading, not compiled, not run on hardware
//
#include <hip/hip_runtime.h>
#include <hip/hip_bf16.h>

#define EMB  768
#define HD   64
#define TSEQ 4096
#define NB   4
#define WSZ  (HD * EMB)          // 49152 elements per weight matrix

typedef __bf16        v16bf __attribute__((ext_vector_type(16)));
typedef float         v8f   __attribute__((ext_vector_type(8)));
typedef unsigned int  u32x4 __attribute__((ext_vector_type(4)));
typedef unsigned int  u32x8 __attribute__((ext_vector_type(8)));

static __device__ __forceinline__ v8f wmma_bf16(v16bf a, v16bf b, v8f c) {
    // D = A(16x32 bf16) * B(32x16 bf16) + C(16x16 f32)
    return __builtin_amdgcn_wmma_f32_16x16x32_bf16(false, a, false, b, (short)0, c, false, false);
}

static __device__ __forceinline__ v16bf concat_bf(u32x4 a, u32x4 b) {
    u32x8 w = __builtin_shufflevector(a, b, 0, 1, 2, 3, 4, 5, 6, 7);
    return __builtin_bit_cast(v16bf, w);
}

static __device__ __forceinline__ v16bf cvt16(float4 a, float4 b, float4 c, float4 d) {
    v16bf v;
    v[0]  = (__bf16)a.x; v[1]  = (__bf16)a.y; v[2]  = (__bf16)a.z; v[3]  = (__bf16)a.w;
    v[4]  = (__bf16)b.x; v[5]  = (__bf16)b.y; v[6]  = (__bf16)b.z; v[7]  = (__bf16)b.w;
    v[8]  = (__bf16)c.x; v[9]  = (__bf16)c.y; v[10] = (__bf16)c.z; v[11] = (__bf16)c.w;
    v[12] = (__bf16)d.x; v[13] = (__bf16)d.y; v[14] = (__bf16)d.z; v[15] = (__bf16)d.w;
    return v;
}

static __device__ __forceinline__ unsigned short bfbits(float f) {
    return __builtin_bit_cast(unsigned short, (__bf16)f);
}

// ---------------------------------------------------------------------------
// Stage 0: one-time fp32 -> bf16 conversion of the three weight matrices.
// Layout: Wb[3][64][768] row-major (same as the source W[out][in]).
// ---------------------------------------------------------------------------
__global__ __launch_bounds__(256) void cvt_w_kernel(
    const float* __restrict__ Wq, const float* __restrict__ Wk,
    const float* __restrict__ Wv, __bf16* __restrict__ Wb)
{
    const int i = blockIdx.x * 256 + threadIdx.x;     // 0 .. 3*WSZ-1
    const float* src = (i < WSZ) ? Wq : ((i < 2 * WSZ) ? Wk : Wv);
    const int off = (i < WSZ) ? i : ((i < 2 * WSZ) ? i - WSZ : i - 2 * WSZ);
    Wb[i] = (__bf16)src[off];
}

// ---------------------------------------------------------------------------
// Stage 1: QKV projection. One wave per 16-token tile (1024 tiles).
//   Q, K -> row-major bf16 [B*T][64]   (Q pre-scaled by 768^-0.5)
//   V    -> transposed bf16 [B][64][T] (so stage-2 B-operand loads are contiguous)
// All 12 B-operands of a k-step are loaded into distinct registers before the
// WMMA chain so the loads pipeline instead of serializing on one register set.
// ---------------------------------------------------------------------------
__global__ __launch_bounds__(128) void qkv_kernel(
    const float* __restrict__ x, const __bf16* __restrict__ Wb,
    __bf16* __restrict__ Qo, __bf16* __restrict__ Ko, __bf16* __restrict__ Vt)
{
    const int wave = threadIdx.x >> 5;
    const int lane = threadIdx.x & 31;
    const int n    = lane & 15;       // column / row-within-tile index
    const int hlf  = lane >> 4;       // which 16-lane half

    const int tile = blockIdx.x * 4 + wave;   // 0..1023
    const int tok0 = tile * 16;               // flat token index (b*T + t)

    v8f acc[12];
    #pragma unroll
    for (int i = 0; i < 12; ++i) acc[i] = (v8f)0.0f;

    const int aoff = hlf * 16;        // A-operand K sub-chunk offset
    const int boff = hlf * 16;        // B-operand K half offset

    const float*  xbase = x  + (size_t)(tok0 + n) * EMB;
    const __bf16* wbase = Wb + (size_t)n * EMB + boff;

    for (int kc = 0; kc < EMB; kc += 32) {
        // Gather all operands for this k-step first (24 B-loads + 4 A-loads
        // issue as clauses; waits stagger across the WMMA chain below).
        v16bf bv[12];
        #pragma unroll
        for (int w = 0; w < 3; ++w) {
            #pragma unroll
            for (int nt = 0; nt < 4; ++nt) {
                const __bf16* wr = wbase + (size_t)w * WSZ + (size_t)(nt * 16) * EMB + kc;
                bv[w * 4 + nt] = concat_bf(*(const u32x4*)wr, *(const u32x4*)(wr + 8));
            }
        }
        const float* xr = xbase + kc + aoff;
        const float4* xv  = (const float4*)xr;
    const float4* xv2 = (const float4*)(xr + 8);
        v16bf av = cvt16(xv[0], xv[1], xv2[0], xv2[1]);
        if (kc + 32 < EMB)
            __builtin_prefetch(xr + 32, 0, 3);    // next k-chunk of this row

        #pragma unroll
        for (int i = 0; i < 12; ++i)
            acc[i] = wmma_bf16(av, bv[i], acc[i]);
    }

    const float scale = 0.03608439182435161f;   // 768^-0.5 (emb_dim per reference)
    const int b    = tok0 / TSEQ;
    const int tloc = tok0 % TSEQ;

    #pragma unroll
    for (int nt = 0; nt < 4; ++nt) {
        const int h = nt * 16 + n;
        const v8f aq = acc[0 * 4 + nt];
        const v8f ak = acc[1 * 4 + nt];
        const v8f avv = acc[2 * 4 + nt];
        // Q / K row-major: accumulator VGPR r holds row (r + 8*hlf), col = h
        #pragma unroll
        for (int r = 0; r < 8; ++r) {
            const size_t tok = (size_t)tok0 + r + hlf * 8;
            Qo[tok * HD + h] = (__bf16)(aq[r] * scale);
            Ko[tok * HD + h] = (__bf16)ak[r];
        }
        // V transposed: lane owns head h; 8 consecutive tokens -> packed 16B store
        unsigned int pk0 = bfbits(avv[0]) | ((unsigned)bfbits(avv[1]) << 16);
        unsigned int pk1 = bfbits(avv[2]) | ((unsigned)bfbits(avv[3]) << 16);
        unsigned int pk2 = bfbits(avv[4]) | ((unsigned)bfbits(avv[5]) << 16);
        unsigned int pk3 = bfbits(avv[6]) | ((unsigned)bfbits(avv[7]) << 16);
        u32x4 pv = {pk0, pk1, pk2, pk3};
        *(u32x4*)(Vt + ((size_t)(b * HD + h)) * TSEQ + tloc + hlf * 8) = pv;
    }
}

// ---------------------------------------------------------------------------
// Stage 2: causal flash attention. One wave per 16-query tile (1024 tiles),
// 32 keys per step, online softmax, per-wave LDS for the P relayout.
// ---------------------------------------------------------------------------
__global__ __launch_bounds__(256) void attn_kernel(
    const __bf16* __restrict__ Q, const __bf16* __restrict__ K,
    const __bf16* __restrict__ Vt, float* __restrict__ out)
{
    __shared__ unsigned short lds[8 * 16 * 32];   // 8 waves x (16 rows x 32 keys)
    const int wave = threadIdx.x >> 5;
    const int lane = threadIdx.x & 31;
    const int n    = lane & 15;
    const int hlf  = lane >> 4;
    unsigned short* pw = lds + wave * 512;

    const int qt = blockIdx.x * 8 + wave;     // 0..1023
    const int b  = qt >> 8;                   // 256 q-tiles per batch
    const int q0 = (qt & 255) * 16;           // first query row (within batch)
    const size_t tokbase = (size_t)b * TSEQ;

    // Q A-operands (K = head dim 64, split into two K=32 chunks)
    const __bf16* qrow = Q + (tokbase + q0 + n) * HD;
    const int aoff = hlf * 16;
    v16bf aq0 = concat_bf(*(const u32x4*)(qrow + aoff),      *(const u32x4*)(qrow + aoff + 8));
    v16bf aq1 = concat_bf(*(const u32x4*)(qrow + 32 + aoff), *(const u32x4*)(qrow + 32 + aoff + 8));

    v8f o[4];
    #pragma unroll
    for (int nt = 0; nt < 4; ++nt) o[nt] = (v8f)0.0f;
    float mrow[8], lrow[8];
    #pragma unroll
    for (int r = 0; r < 8; ++r) { mrow[r] = -1e30f; lrow[r] = 0.0f; }

    const v8f zc = (v8f)0.0f;
    const int kgmax = (q0 + 15) >> 5;          // last 32-key group touching the diagonal
    const __bf16* vallbase = Vt + ((size_t)b * HD) * TSEQ + hlf * 16;

    for (int kg = 0; kg <= kgmax; ++kg) {
        const int key0 = kg * 32;

        // K B-operands: lane = key column, K-contiguous head dims (row of K)
        const __bf16* kr0 = K + (tokbase + key0 + n) * HD + hlf * 16;
        const __bf16* kr1 = kr0 + 16 * HD;
        // V B-operands: lane = head column, K-contiguous keys (V stored transposed)
        const __bf16* vbase = vallbase + key0;
        if (kg < kgmax) {                         // warm next key group
            __builtin_prefetch(kr0 + 32 * HD, 0, 3);
            __builtin_prefetch(kr1 + 32 * HD, 0, 3);
            __builtin_prefetch(vbase + (size_t)n * TSEQ + 32, 0, 3);
        }
        v16bf kb00 = concat_bf(*(const u32x4*)kr0,        *(const u32x4*)(kr0 + 8));
        v16bf kb01 = concat_bf(*(const u32x4*)(kr0 + 32), *(const u32x4*)(kr0 + 40));
        v16bf kb10 = concat_bf(*(const u32x4*)kr1,        *(const u32x4*)(kr1 + 8));
        v16bf kb11 = concat_bf(*(const u32x4*)(kr1 + 32), *(const u32x4*)(kr1 + 40));
        v16bf vb[4];
        #pragma unroll
        for (int nt = 0; nt < 4; ++nt) {
            const __bf16* vr = vbase + (size_t)(nt * 16 + n) * TSEQ;
            vb[nt] = concat_bf(*(const u32x4*)vr, *(const u32x4*)(vr + 8));
        }

        // S = Q * K^T  (scale already folded into Q)
        v8f s0 = wmma_bf16(aq0, kb00, zc); s0 = wmma_bf16(aq1, kb01, s0);
        v8f s1 = wmma_bf16(aq0, kb10, zc); s1 = wmma_bf16(aq1, kb11, s1);

        // causal mask (uniform branch per wave)
        if (key0 + 31 > q0) {
            #pragma unroll
            for (int r = 0; r < 8; ++r) {
                const int rowg = q0 + r + hlf * 8;
                if (key0 + n      > rowg) s0[r] = -1e30f;
                if (key0 + 16 + n > rowg) s1[r] = -1e30f;
            }
        }

        // online softmax: reduce over the 16 key-columns (lanes within each half)
        #pragma unroll
        for (int r = 0; r < 8; ++r) {
            float mx = fmaxf(s0[r], s1[r]);
            mx = fmaxf(mx, __shfl_xor(mx, 1));
            mx = fmaxf(mx, __shfl_xor(mx, 2));
            mx = fmaxf(mx, __shfl_xor(mx, 4));
            mx = fmaxf(mx, __shfl_xor(mx, 8));
            const float mnew  = fmaxf(mrow[r], mx);
            const float alpha = __expf(mrow[r] - mnew);
            mrow[r] = mnew;
            const float p0 = __expf(s0[r] - mnew);
            const float p1 = __expf(s1[r] - mnew);
            float sum = p0 + p1;
            sum += __shfl_xor(sum, 1);
            sum += __shfl_xor(sum, 2);
            sum += __shfl_xor(sum, 4);
            sum += __shfl_xor(sum, 8);
            lrow[r] = lrow[r] * alpha + sum;
            #pragma unroll
            for (int nt = 0; nt < 4; ++nt) o[nt][r] *= alpha;
            s0[r] = p0; s1[r] = p1;
        }

        // P -> LDS (C layout) -> reload in A layout (same-wave dscnt dependency)
        #pragma unroll
        for (int r = 0; r < 8; ++r) {
            const int row = r + hlf * 8;
            pw[row * 32 + n]      = bfbits(s0[r]);
            pw[row * 32 + 16 + n] = bfbits(s1[r]);
        }
        const unsigned short* pr = pw + n * 32;
        v16bf pa = concat_bf(*(const u32x4*)(pr + aoff), *(const u32x4*)(pr + aoff + 8));

        // O += P * V
        #pragma unroll
        for (int nt = 0; nt < 4; ++nt)
            o[nt] = wmma_bf16(pa, vb[nt], o[nt]);
    }

    // epilogue: O / l, fp32 row-major
    float* orow = out + (tokbase + q0) * HD;
    #pragma unroll
    for (int r = 0; r < 8; ++r) {
        const float inv = 1.0f / lrow[r];
        const int row = r + hlf * 8;
        #pragma unroll
        for (int nt = 0; nt < 4; ++nt)
            orow[(size_t)row * HD + nt * 16 + n] = o[nt][r] * inv;
    }
}

extern "C" void kernel_launch(void* const* d_in, const int* in_sizes, int n_in,
                              void* d_out, int out_size, void* d_ws, size_t ws_size,
                              hipStream_t stream) {
    const float* x  = (const float*)d_in[0];
    const float* Wq = (const float*)d_in[1];
    const float* Wk = (const float*)d_in[2];
    const float* Wv = (const float*)d_in[3];

    const size_t ntok = (size_t)NB * TSEQ;
    __bf16* Qb = (__bf16*)d_ws;
    __bf16* Kb = Qb + ntok * HD;
    __bf16* Vt = Kb + ntok * HD;          // [B][64][T]
    __bf16* Wb = Vt + ntok * HD;          // [3][64][768] bf16

    // one-time weight conversion (3*49152 elements)
    cvt_w_kernel<<<(3 * WSZ) / 256, 256, 0, stream>>>(Wq, Wk, Wv, Wb);
    // 1024 16-token tiles, 4 waves/block
    qkv_kernel<<<256, 128, 0, stream>>>(x, Wb, Qb, Kb, Vt);
    // 1024 16-query tiles, 8 waves/block
    attn_kernel<<<128, 256, 0, stream>>>(Qb, Kb, Vt, (float*)d_out);
}
